// EffAttention_43937515438675
// MI455X (gfx1250) — matrix operand order repaired and checked by reading, not compiled, not run on hardware
//
#include <hip/hip_runtime.h>
#include <hip/hip_bf16.h>

typedef __attribute__((ext_vector_type(16))) __bf16 v16bf;
typedef __attribute__((ext_vector_type(8)))  __bf16 v8bf;
typedef __attribute__((ext_vector_type(8)))  float  v8f;

#define SCALE_F 0.08838834764831845f  // (1024/8)^-0.5

#if defined(__AMDGCN__) && __has_builtin(__builtin_amdgcn_global_load_async_to_lds_b128)
#define HAVE_ASYNC_LDS 1
#else
#define HAVE_ASYNC_LDS 0
#endif

// ---------- helpers ----------
__device__ __forceinline__ unsigned short f2bf(float f) {
    unsigned int u; __builtin_memcpy(&u, &f, 4);
    unsigned int r = u + 0x7FFFu + ((u >> 16) & 1u);   // round-to-nearest-even
    return (unsigned short)(r >> 16);
}
__device__ __forceinline__ float ldf(const float* p) { return *p; }
__device__ __forceinline__ float ldf(const unsigned short* p) {
    unsigned int u = ((unsigned int)(*p)) << 16;
    float f; __builtin_memcpy(&f, &u, 4); return f;
}
// A-fragment: two contiguous 8x bf16 chunks (K 16h..16h+7 / K 16h+8..16h+15 for half h)
__device__ __forceinline__ v16bf make_frag(const unsigned short* lo, const unsigned short* hi) {
    v8bf a = *(const v8bf*)lo;
    v8bf b = *(const v8bf*)hi;
    v16bf r;
#pragma unroll
    for (int i = 0; i < 8; ++i) { r[i] = a[i]; r[i + 8] = b[i]; }
    return r;
}
__device__ __forceinline__ void st_out(float* p, float v)          { *p = v; }
__device__ __forceinline__ void st_out(unsigned short* p, float v) { *p = f2bf(v); }

#if HAVE_ASYNC_LDS
// 16-byte global -> LDS async copy (tracked on ASYNCcnt, no VGPR data)
typedef int v4i_vec __attribute__((vector_size(16)));
typedef __attribute__((address_space(1))) v4i_vec* g_v4i_p;
typedef __attribute__((address_space(3))) v4i_vec* l_v4i_p;
__device__ __forceinline__ void async_cp16(const void* g, void* l) {
    __builtin_amdgcn_global_load_async_to_lds_b128((g_v4i_p)(g), (l_v4i_p)(l), 0, 0);
}
__device__ __forceinline__ void async_wait0() {
#if __has_builtin(__builtin_amdgcn_s_wait_asynccnt)
    __builtin_amdgcn_s_wait_asynccnt(0);
#else
    asm volatile("s_wait_asynccnt 0x0" ::: "memory");
#endif
}
#endif

// ---------- generic bf16-WMMA GEMM: C[M,N] = A[M,K] * B[K,N] (+bias) ----------
// Block tile 128x128, K-step 32. 8 waves: 2x4 grid, each wave 64x32 (4x2 WMMA tiles).
template <typename AT, typename OT, bool HAS_BIAS>
__global__ __launch_bounds__(256) void gemm_wmma_bf16(const AT* __restrict__ A,
                                                      const float* __restrict__ B,
                                                      OT* __restrict__ C,
                                                      const float* __restrict__ bias,
                                                      int M, int N, int K) {
    __shared__ __align__(16) unsigned short As[128][32];   // [m][k]  (K contiguous)
    __shared__ __align__(16) unsigned short Bs[128][32];   // [n][k]  (transposed: K contiguous)

    const int tid   = threadIdx.x;
    const int nbase = blockIdx.x * 128;
    const int mbase = blockIdx.y * 128;
    const int w     = tid >> 5;
    const int lane  = tid & 31;
    const int wrow  = w >> 2;        // 0..1
    const int wcol  = w & 3;         // 0..3
    const int lrow  = lane & 15;
    const int hf    = lane >> 4;     // wave half selects K/M sub-ranges per ISA layout

    v8f acc[4][2];
#pragma unroll
    for (int mt = 0; mt < 4; ++mt)
#pragma unroll
        for (int nt = 0; nt < 2; ++nt) { v8f z = {}; acc[mt][nt] = z; }

    for (int k0 = 0; k0 < K; k0 += 32) {
        // ---- stage A tile ----
        bool used_async = false;
        if constexpr (sizeof(AT) == 2) {
#if HAVE_ASYNC_LDS
            // bf16 source, layout-preserving copy: async DMA, 16B per op, 2 ops/thread
#pragma unroll
            for (int i = 0; i < 2; ++i) {
                int e   = tid + i * 256;
                int r   = e >> 2;
                int seg = e & 3;
                async_cp16(A + (size_t)(mbase + r) * K + k0 + seg * 8, &As[r][seg * 8]);
            }
            used_async = true;
#endif
        }
        if (!used_async) {
            // fp32 (or fallback) source: convert to bf16 through VGPRs
#pragma unroll
            for (int i = 0; i < 4; ++i) {
                int e  = tid + i * 256;
                int r  = e >> 3;
                int kq = (e & 7) * 4;
                const AT* src = A + (size_t)(mbase + r) * K + k0 + kq;
#pragma unroll
                for (int j = 0; j < 4; ++j) As[r][kq + j] = f2bf(ldf(src + j));
            }
        }
        // ---- stage B tile transposed: Bs[n][k] (convert fp32 -> bf16) ----
#pragma unroll
        for (int i = 0; i < 16; ++i) {
            int e  = tid + i * 256;
            int kk = e >> 7;
            int n  = e & 127;
            Bs[n][kk] = f2bf(B[(size_t)(k0 + kk) * N + nbase + n]);
        }
        // prefetch next K-tile of B into cache (global_prefetch_b8)
        if (k0 + 32 < K) {
            int kk = tid >> 3, n = (tid & 7) * 16;
            __builtin_prefetch(B + (size_t)(k0 + 32 + kk) * N + nbase + n, 0, 0);
        }
#if HAVE_ASYNC_LDS
        if (used_async) async_wait0();
#endif
        __syncthreads();

        v16bf afrag[4], bfrag[2];
#pragma unroll
        for (int mt = 0; mt < 4; ++mt) {
            int r  = wrow * 64 + mt * 16 + lrow;
            int kb = hf * 16;
            afrag[mt] = make_frag(&As[r][kb], &As[r][kb + 8]);
        }
#pragma unroll
        for (int nt = 0; nt < 2; ++nt) {
            int n  = wcol * 32 + nt * 16 + lrow;
            int kb = hf * 16;                      // lanes 0-15: K0..15, lanes 16-31: K16..31
            bfrag[nt] = *(const v16bf*)&Bs[n][kb]; // 32B aligned, contiguous
        }
#pragma unroll
        for (int mt = 0; mt < 4; ++mt)
#pragma unroll
            for (int nt = 0; nt < 2; ++nt)
                acc[mt][nt] = __builtin_amdgcn_wmma_f32_16x16x32_bf16(
                    false, afrag[mt], false, bfrag[nt], (short)0, acc[mt][nt], false, false);
        __syncthreads();
    }

    // epilogue: C/D layout -> global
#pragma unroll
    for (int mt = 0; mt < 4; ++mt)
#pragma unroll
        for (int nt = 0; nt < 2; ++nt) {
            int n = nbase + wcol * 32 + nt * 16 + lrow;
            float bv = HAS_BIAS ? bias[n] : 0.0f;
#pragma unroll
            for (int r = 0; r < 8; ++r) {
                int m = mbase + wrow * 64 + mt * 16 + r + hf * 8;
                st_out(&C[(size_t)m * N + n], acc[mt][nt][r] + bv);
            }
        }
}

// ---------- block-local flash attention ----------
// qkv: bf16 [B*N, 1536], columns (which*512 + h*64 + d). out: bf16 [B*N, 512], col h*64+d.
// Grid: b*H*chunks*(1024/128) blocks; each block: 128 q rows, stream k/v in 64-row tiles.
__global__ __launch_bounds__(256) void attn_wmma_bf16(const unsigned short* __restrict__ qkv,
                                                      unsigned short* __restrict__ out) {
    __shared__ __align__(16) unsigned short qs[128][64];   // [qrow][d]
    __shared__ __align__(16) unsigned short ks[64][64];    // [krow][d] (natural = kT B-frag layout)
    __shared__ __align__(16) unsigned short vs[64][64];    // [d][krow] (transposed for P*V)
    __shared__ __align__(16) unsigned short ps[8][16][64]; // per-wave P staging (C->A relayout)

    const int tid  = threadIdx.x;
    const int w    = tid >> 5;
    const int lane = tid & 31;
    const int lrow = lane & 15;
    const int hf   = lane >> 4;

    int bid   = blockIdx.x;
    int qb    = bid & 7;  bid >>= 3;   // 128-row sub-block within chunk
    int chunk = bid & 3;  bid >>= 2;
    int h     = bid & 7;  bid >>= 3;
    int b     = bid;

    const int    qrow0  = chunk * 1024 + qb * 128;
    const size_t baseBN = (size_t)b * 4096;
    const int colq = h * 64, colk = 512 + h * 64, colv = 1024 + h * 64;

    // ---- stage q rows once (layout-preserving bf16 copy) ----
#if HAVE_ASYNC_LDS
#pragma unroll
    for (int i = 0; i < 4; ++i) {
        int e   = tid + i * 256;
        int r   = e >> 3;
        int seg = e & 7;
        async_cp16(&qkv[(baseBN + qrow0 + r) * 1536 + colq + seg * 8], &qs[r][seg * 8]);
    }
#else
#pragma unroll
    for (int i = 0; i < 32; ++i) {
        int e = tid + i * 256;
        int r = e >> 6, d = e & 63;
        qs[r][d] = qkv[(baseBN + qrow0 + r) * 1536 + colq + d];
    }
#endif

    float m_state[8], l_state[8];
    v8f acc_o[4];
#pragma unroll
    for (int r = 0; r < 8; ++r) { m_state[r] = -1e30f; l_state[r] = 0.0f; }
#pragma unroll
    for (int dt = 0; dt < 4; ++dt) { v8f z = {}; acc_o[dt] = z; }

    for (int kt = 0; kt < 16; ++kt) {
        __syncthreads();   // previous tile reads done
        const int krow0 = chunk * 1024 + kt * 64;
        // ---- k tile: layout-preserving copy -> async; v tile: transpose -> VGPR path ----
#if HAVE_ASYNC_LDS
#pragma unroll
        for (int i = 0; i < 2; ++i) {
            int e   = tid + i * 256;
            int r   = e >> 3;
            int seg = e & 7;
            async_cp16(&qkv[(baseBN + krow0 + r) * 1536 + colk + seg * 8], &ks[r][seg * 8]);
        }
#endif
#pragma unroll
        for (int i = 0; i < 16; ++i) {
            int e = tid + i * 256;
            int r = e >> 6, d = e & 63;
            size_t rowoff = (baseBN + krow0 + r) * 1536;
#if !HAVE_ASYNC_LDS
            ks[r][d] = qkv[rowoff + colk + d];
#endif
            vs[d][r] = qkv[rowoff + colv + d];
        }
#if HAVE_ASYNC_LDS
        async_wait0();     // k tile (and, iter 0, q tile) landed in LDS
#endif
        __syncthreads();

        // S = q * k^T  (M=16 per wave, N=64, K=d=64)
        const int rowb = w * 16;
        v8f s[4];
#pragma unroll
        for (int nt = 0; nt < 4; ++nt) { v8f z = {}; s[nt] = z; }
#pragma unroll
        for (int koff = 0; koff < 64; koff += 32) {
            int kb = hf * 16 + koff;
            v16bf a = make_frag(&qs[rowb + lrow][kb], &qs[rowb + lrow][kb + 8]);
#pragma unroll
            for (int nt = 0; nt < 4; ++nt) {
                v16bf bf = *(const v16bf*)&ks[nt * 16 + lrow][koff + hf * 16];
                s[nt] = __builtin_amdgcn_wmma_f32_16x16x32_bf16(
                    false, a, false, bf, (short)0, s[nt], false, false);
            }
        }

        // online softmax update (rows = VGPR idx + 8*half; cols spread over 16-lane halves)
#pragma unroll
        for (int r = 0; r < 8; ++r) {
            float mx = -1e30f;
#pragma unroll
            for (int nt = 0; nt < 4; ++nt) { float v = s[nt][r] * SCALE_F; s[nt][r] = v; mx = fmaxf(mx, v); }
#pragma unroll
            for (int off = 1; off < 16; off <<= 1) mx = fmaxf(mx, __shfl_xor(mx, off, 32));
            float nm = fmaxf(m_state[r], mx);
            float cf = __expf(m_state[r] - nm);
            m_state[r] = nm;
            float rs = 0.0f;
#pragma unroll
            for (int nt = 0; nt < 4; ++nt) { float p = __expf(s[nt][r] - nm); s[nt][r] = p; rs += p; }
#pragma unroll
            for (int off = 1; off < 16; off <<= 1) rs += __shfl_xor(rs, off, 32);
            l_state[r] = l_state[r] * cf + rs;
#pragma unroll
            for (int dt = 0; dt < 4; ++dt) acc_o[dt][r] *= cf;
            // C-layout -> LDS staging for A-fragment reuse
            int prow = r + hf * 8;
#pragma unroll
            for (int nt = 0; nt < 4; ++nt) ps[w][prow][nt * 16 + lrow] = f2bf(s[nt][r]);
        }
        __builtin_amdgcn_wave_barrier();   // same-wave DS ops are in-order; pin schedule

        // O += P * V  (K = 64 k-rows, N = d = 64)
#pragma unroll
        for (int koff = 0; koff < 64; koff += 32) {
            int kb = hf * 16 + koff;
            v16bf a = make_frag(&ps[w][lrow][kb], &ps[w][lrow][kb + 8]);
#pragma unroll
            for (int dt = 0; dt < 4; ++dt) {
                v16bf bf = *(const v16bf*)&vs[dt * 16 + lrow][koff + hf * 16];
                acc_o[dt] = __builtin_amdgcn_wmma_f32_16x16x32_bf16(
                    false, a, false, bf, (short)0, acc_o[dt], false, false);
            }
        }
    }

    // normalize + store [B*N, 512] bf16
#pragma unroll
    for (int r = 0; r < 8; ++r) {
        float inv = 1.0f / l_state[r];
        int gm = qrow0 + w * 16 + r + hf * 8;
#pragma unroll
        for (int dt = 0; dt < 4; ++dt) {
            int c = h * 64 + dt * 16 + lrow;
            out[(baseBN + gm) * 512 + c] = f2bf(acc_o[dt][r] * inv);
        }
    }
}

// ---------- launch ----------
extern "C" void kernel_launch(void* const* d_in, const int* in_sizes, int n_in,
                              void* d_out, int out_size, void* d_ws, size_t ws_size,
                              hipStream_t stream) {
    const float* x        = (const float*)d_in[0];   // [4,4096,1024]
    const float* w_reduce = (const float*)d_in[1];   // [1024,512]
    const float* w_qkv    = (const float*)d_in[2];   // [512,1536]
    const float* w_proj   = (const float*)d_in[3];   // [512,1024]
    const float* b_proj   = (const float*)d_in[4];   // [1024]
    float*       outp     = (float*)d_out;           // [4,4096,1024]

    const int M = 4 * 4096;                          // 16384 rows
    unsigned short* xr  = (unsigned short*)d_ws;                 // bf16 [M,512]  (16 MB)
    unsigned short* qkv = xr + (size_t)M * 512;                  // bf16 [M,1536] (48 MB)
    unsigned short* att = xr;                                    // reuse xr region after GEMM2

    dim3 blk(256);
    // 1) xr = x @ w_reduce          (M=16384, N=512, K=1024)
    gemm_wmma_bf16<float, unsigned short, false>
        <<<dim3(512 / 128, M / 128), blk, 0, stream>>>(x, w_reduce, xr, nullptr, M, 512, 1024);
    // 2) qkv = xr @ w_qkv           (M=16384, N=1536, K=512)
    gemm_wmma_bf16<unsigned short, unsigned short, false>
        <<<dim3(1536 / 128, M / 128), blk, 0, stream>>>(xr, w_qkv, qkv, nullptr, M, 1536, 512);
    // 3) block-local attention      (4*8*4 chunks * 8 sub-blocks = 1024 blocks)
    attn_wmma_bf16<<<dim3(1024), blk, 0, stream>>>(qkv, att);
    // 4) out = att @ w_proj + b     (M=16384, N=1024, K=512)
    gemm_wmma_bf16<unsigned short, float, true>
        <<<dim3(1024 / 128, M / 128), blk, 0, stream>>>(att, w_proj, outp, b_proj, M, 1024, 512);
}
